// modified_transformer_9440338117007
// MI455X (gfx1250) — compile-verified, matrix operand order checked by reading, not run on hardware
//
#include <hip/hip_runtime.h>
#include <hip/hip_bf16.h>
#include <stdint.h>

// Problem dims (fixed by reference)
#define B_ 16
#define T_ 512
#define F_ 512
#define H_ 8
#define D_ 64
#define ROWS_ (B_ * T_)   // 8192

typedef __bf16 bf16;
typedef __attribute__((ext_vector_type(16))) __bf16 v16bf;
typedef __attribute__((ext_vector_type(8)))  float  v8f;

#define WMMA_BF16(a, b, c) \
  __builtin_amdgcn_wmma_f32_16x16x32_bf16(false, (a), false, (b), (short)0, (c), false, false)

union FragU { uint4 u[2]; v16bf v; };

// Load one 16x32 bf16 fragment (A row-major, or B as Bt[N,K] row-major).
// Lane l (0..15) = row/col l; K layout per CDNA5 ISA: VGPR0-3 hold K {0..7} (lanes 0-15)
// / {8..15} (lanes 16-31), VGPR4-7 hold K+16..  => two contiguous 16B loads per lane.
__device__ inline v16bf load_frag(const bf16* base, int stride) {
  int lane = threadIdx.x & 31;
  const bf16* p = base + (size_t)(lane & 15) * stride + ((lane & 16) ? 8 : 0);
  FragU f;
  f.u[0] = *(const uint4*)(p);
  f.u[1] = *(const uint4*)(p + 16);
  return f.v;
}

// ---------------------------------------------------------------------------
// CDNA5 async global->LDS copy (16B per lane), tracked by ASYNCcnt.
// VDST operand is a VGPR holding the LDS byte offset (flat aperture: addr[31:0]).
// ---------------------------------------------------------------------------
__device__ inline void async_ld16(void* lds_dst, const void* gsrc) {
  uint32_t loff = (uint32_t)(uintptr_t)lds_dst;
  uint64_t g = (uint64_t)(uintptr_t)gsrc;
  asm volatile("global_load_async_to_lds_b128 %0, %1, off" :: "v"(loff), "v"(g) : "memory");
}

__device__ inline void wait_async0() {
#if __has_builtin(__builtin_amdgcn_s_wait_asynccnt)
  __builtin_amdgcn_s_wait_asynccnt(0);
#else
  asm volatile("s_wait_asynccnt 0" ::: "memory");
#endif
}

// ---------------------------------------------------------------------------
// Elementwise / prep kernels
// ---------------------------------------------------------------------------
__global__ void k_init(float* h0, float* h1, unsigned* cnt) {
  int i = blockIdx.x * blockDim.x + threadIdx.x;
  if (i < B_ * F_) { h0[i] = 0.f; h1[i] = 0.f; }
  if (i == 0) *cnt = 0u;
}

__global__ void k_f32_to_bf16(const float* __restrict__ S, bf16* __restrict__ D, size_t n) {
  for (size_t i = (size_t)blockIdx.x * blockDim.x + threadIdx.x; i < n;
       i += (size_t)gridDim.x * blockDim.x)
    D[i] = (bf16)S[i];
}

// D[c*R + r] = S[r*C + c]  (f32 [R,C] -> bf16 [C,R])
__global__ void k_transpose_to_bf16(const float* __restrict__ S, bf16* __restrict__ D,
                                    int R, int C) {
  size_t n = (size_t)R * C;
  for (size_t i = (size_t)blockIdx.x * blockDim.x + threadIdx.x; i < n;
       i += (size_t)gridDim.x * blockDim.x) {
    size_t r = i / C, c = i % C;
    D[c * (size_t)R + r] = (bf16)S[i];
  }
}

// V f32 [B,T,H,D] -> VT bf16 [B,H,D,T]
__global__ void k_transpose_v(const float* __restrict__ V, bf16* __restrict__ VT) {
  size_t n = (size_t)ROWS_ * F_;
  for (size_t i = (size_t)blockIdx.x * blockDim.x + threadIdx.x; i < n;
       i += (size_t)gridDim.x * blockDim.x) {
    int d = (int)(i % D_);
    int h = (int)((i / D_) % H_);
    int t = (int)((i / (D_ * H_)) % T_);
    int b = (int)(i / ((size_t)D_ * H_ * T_));
    VT[(((size_t)b * H_ + h) * D_ + d) * T_ + t] = (bf16)V[i];
  }
}

// ---------------------------------------------------------------------------
// Generic bf16 WMMA GEMM: C[M,N] (f32) = A[M,K] * Bt[N,K]^T + bias[N]
// Block = 256 thr (8 waves); wave -> 16(M) x 64(N); grid = (N/64, M/128).
// B tile staged in LDS via async DMA (double-buffered, overlapped with WMMA);
// A fragment register double-buffered.
// ---------------------------------------------------------------------------
__global__ __launch_bounds__(256) void k_gemm_bf16(
    const bf16* __restrict__ A, const bf16* __restrict__ Bt,
    const float* __restrict__ bias, float* __restrict__ C,
    int M, int N, int K) {
  __shared__ __align__(16) bf16 Bs[2][64 * 32];  // 4 KB per buffer
  int tid = threadIdx.x;
  int lane = tid & 31;
  int wave = tid >> 5;
  int m0 = blockIdx.y * 128 + wave * 16;
  int n0 = blockIdx.x * 64;

  // staging map: 256 threads x 16B = 4KB chunk (64 rows x 32 k, row stride 32)
  int srow = tid >> 2, sq = tid & 3;
  const bf16* bsrc = Bt + (size_t)(n0 + srow) * K + sq * 8;
  bf16* bdst[2] = { &Bs[0][srow * 32 + sq * 8], &Bs[1][srow * 32 + sq * 8] };

  const bf16* Arow = A + (size_t)m0 * K;
  v8f acc[4] = {{}, {}, {}, {}};

  // prologue: stage k=0 into buf0, prefetch first A fragment
  async_ld16(bdst[0], bsrc);
  v16bf afrag = load_frag(Arow, K);
  wait_async0();
  __syncthreads();

  for (int k = 0; k < K; k += 32) {
    int buf = (k >> 5) & 1;
    bool more = (k + 32) < K;
    if (more) async_ld16(bdst[buf ^ 1], bsrc + (k + 32));  // DMA next chunk
    v16bf afnext = afrag;
    if (more) afnext = load_frag(Arow + k + 32, K);        // prefetch next A
    const bf16* bb = &Bs[buf][0];
#pragma unroll
    for (int nt = 0; nt < 4; ++nt) {
      v16bf bfr = load_frag(bb + nt * 16 * 32, 32);
      acc[nt] = WMMA_BF16(afrag, bfr, acc[nt]);
    }
    afrag = afnext;
    wait_async0();      // next chunk landed (overlapped with the 4 wmma above)
    __syncthreads();
  }

  int nc = lane & 15;
  int mhi = (lane & 16) ? 8 : 0;
#pragma unroll
  for (int nt = 0; nt < 4; ++nt) {
    int n = n0 + nt * 16 + nc;
    float bv = bias[n];
#pragma unroll
    for (int v = 0; v < 8; ++v)
      C[(size_t)(m0 + mhi + v) * N + n] = acc[nt][v] + bv;
  }
}

// ---------------------------------------------------------------------------
// Attention: one wave per (b, h, 16-query tile). WMMA scores -> LDS softmax
// -> WMMA context (against pre-transposed V). Output ctx in bf16 [B,T,H*D].
// ---------------------------------------------------------------------------
__global__ __launch_bounds__(32) void k_attn(
    const bf16* __restrict__ Q, const bf16* __restrict__ Kb,
    const bf16* __restrict__ VT, bf16* __restrict__ CTX) {
  __shared__ __align__(16) float Ssc[16 * T_];  // scores f32 (32 KB)
  __shared__ __align__(16) bf16  P[16 * T_];    // attn bf16 (16 KB)
  int lane = threadIdx.x;
  int t0 = blockIdx.x * 16;
  int h  = blockIdx.y;
  int b  = blockIdx.z;
  const bf16* qb = Q  + ((size_t)b * T_ + t0) * F_ + h * D_;
  const bf16* kb = Kb + (size_t)b * T_ * F_ + h * D_;
  int nc = lane & 15;
  int mhi = (lane & 16) ? 8 : 0;

  // scores = q k^T / 8  (K = D = 64 -> 2 wmma per 16x16 tile)
  for (int s0 = 0; s0 < T_; s0 += 16) {
    v8f acc = {};
    for (int k = 0; k < D_; k += 32) {
      v16bf af  = load_frag(qb + k, F_);
      v16bf bfr = load_frag(kb + (size_t)s0 * F_ + k, F_);
      acc = WMMA_BF16(af, bfr, acc);
    }
#pragma unroll
    for (int v = 0; v < 8; ++v)
      Ssc[(mhi + v) * T_ + s0 + nc] = acc[v] * 0.125f;
  }
  __syncthreads();

  // softmax: 2 lanes per row, 256 cols each
  {
    int row = lane >> 1;
    int c0 = (lane & 1) * 256;
    float mx = -3.4e38f;
    for (int c = 0; c < 256; ++c) mx = fmaxf(mx, Ssc[row * T_ + c0 + c]);
    mx = fmaxf(mx, __shfl_xor(mx, 1));
    float sum = 0.f;
    for (int c = 0; c < 256; ++c) {
      float e = __expf(Ssc[row * T_ + c0 + c] - mx);
      Ssc[row * T_ + c0 + c] = e;
      sum += e;
    }
    sum += __shfl_xor(sum, 1);
    float inv = 1.f / sum;
    for (int c = 0; c < 256; ++c)
      P[row * T_ + c0 + c] = (bf16)(Ssc[row * T_ + c0 + c] * inv);
  }
  __syncthreads();

  // ctx = attn @ v : A = P (16 x 512), Bt = VT[b,h] (64 rows x 512)
  const bf16* vtb = VT + ((size_t)b * H_ + h) * D_ * T_;
  v8f acc[4] = {{}, {}, {}, {}};
  for (int k = 0; k < T_; k += 32) {
    v16bf af = load_frag(&P[k], T_);
#pragma unroll
    for (int nt = 0; nt < 4; ++nt) {
      v16bf bfr = load_frag(vtb + (size_t)(nt * 16) * T_ + k, T_);
      acc[nt] = WMMA_BF16(af, bfr, acc[nt]);
    }
  }
#pragma unroll
  for (int nt = 0; nt < 4; ++nt)
#pragma unroll
    for (int v = 0; v < 8; ++v)
      CTX[((size_t)b * T_ + t0 + mhi + v) * F_ + h * D_ + nt * 16 + nc] =
          (bf16)acc[nt][v];
}

// ---------------------------------------------------------------------------
// LayerNorm(a + res) * g + b over F=512; optional bf16 mirror output
// ---------------------------------------------------------------------------
__global__ __launch_bounds__(256) void k_layernorm(
    const float* __restrict__ A, const float* __restrict__ R,
    const float* __restrict__ G, const float* __restrict__ Bv,
    float* __restrict__ O, bf16* __restrict__ Obf) {
  int row = blockIdx.x;
  int tid = threadIdx.x;
  int wave = tid >> 5, lane = tid & 31;
  __shared__ float sb[8], qb[8];
  __shared__ float mu_s, inv_s;
  const float* pa = A + (size_t)row * F_;
  const float* pr = R + (size_t)row * F_;
  float v0 = pa[tid] + pr[tid];
  float v1 = pa[tid + 256] + pr[tid + 256];
  float s = v0 + v1, q = v0 * v0 + v1 * v1;
  for (int off = 16; off > 0; off >>= 1) {
    s += __shfl_xor(s, off);
    q += __shfl_xor(q, off);
  }
  if (lane == 0) { sb[wave] = s; qb[wave] = q; }
  __syncthreads();
  if (tid == 0) {
    float S = 0.f, Qq = 0.f;
    for (int i = 0; i < 8; ++i) { S += sb[i]; Qq += qb[i]; }
    float mu = S / F_;
    mu_s = mu;
    inv_s = rsqrtf(Qq / F_ - mu * mu + 1e-3f);
  }
  __syncthreads();
  float mu = mu_s, inv = inv_s;
  float o0 = (v0 - mu) * inv * G[tid] + Bv[tid];
  float o1 = (v1 - mu) * inv * G[tid + 256] + Bv[tid + 256];
  O[(size_t)row * F_ + tid] = o0;
  O[(size_t)row * F_ + tid + 256] = o1;
  if (Obf) {
    Obf[(size_t)row * F_ + tid] = (bf16)o0;
    Obf[(size_t)row * F_ + tid + 256] = (bf16)o1;
  }
}

// ---------------------------------------------------------------------------
// GRU recurrent scan. 16 cooperative workgroups; each owns 32 output features
// = 96 of the 1536 recurrent columns. Its 96x512 bf16 weight slice (96 KB) is
// preloaded ONCE into LDS via async DMA and stays resident for all 512 steps
// (CDNA5's 320 KB LDS). h kept in LDS (f32 + bf16); recurrent matmul is bf16
// WMMA fed entirely from LDS. Steps separated by monotonic atomic grid barrier.
// ---------------------------------------------------------------------------
__global__ __launch_bounds__(256) void k_gru_scan(
    const float* __restrict__ XP,      // [B*T, 1536] input projections + bias
    const bf16* __restrict__ RKt,      // [1536, 512] recurrent weights^T bf16
    const float* __restrict__ rbias,   // [1536] recurrent bias
    float* __restrict__ h0buf, float* __restrict__ h1buf,  // [16,512] each
    bf16* __restrict__ Gbf,            // [B*T, 512] output (bf16 for Dense)
    unsigned* __restrict__ cnt) {
  __shared__ __align__(16) bf16 Bsl[96 * F_];  // 96 KB: resident weight slice
  __shared__ __align__(16) bf16 hbf[B_ * F_];  // 16 KB
  __shared__ float hf[B_ * F_];                // 32 KB
  __shared__ float rec[B_ * 96];               // 6 KB
  int tid = threadIdx.x, lane = tid & 31, wave = tid >> 5;
  int g = blockIdx.x;                 // feature group: owns f in [g*32, g*32+32)
  unsigned nb = gridDim.x;

  // one-time async preload of the weight slice: local row lr <-> global row
  // grow = gate*512 + g*32 + (lr&31), gate = lr>>5. 96 rows x 64 x 16B chunks.
  for (int idx = tid; idx < 96 * 64; idx += 256) {
    int lr = idx >> 6, c = idx & 63;
    int grow = (lr >> 5) * F_ + g * 32 + (lr & 31);
    async_ld16(&Bsl[(size_t)lr * F_ + c * 8], RKt + (size_t)grow * F_ + c * 8);
  }
  wait_async0();
  __syncthreads();

  for (int t = 0; t < T_; ++t) {
    const float* hcur = (t & 1) ? h1buf : h0buf;
    float* hnxt = (t & 1) ? h0buf : h1buf;

    // phase A: snapshot h into LDS (f32 + bf16)
    for (int i = tid; i < B_ * F_; i += 256) {
      float hv = hcur[i];
      hf[i] = hv;
      hbf[i] = (bf16)hv;
    }
    __syncthreads();

    // phase B: rec = h @ RK (our 96 columns). 6 n-tiles on waves 0..5,
    // both operands from LDS.
    if (wave < 6) {
      const bf16* bbase = Bsl + (size_t)(wave * 16) * F_;
      v8f acc = {};
      for (int k = 0; k < F_; k += 32) {
        v16bf af  = load_frag(hbf + k, F_);
        v16bf bfr = load_frag(bbase + k, F_);
        acc = WMMA_BF16(af, bfr, acc);
      }
      int nc = lane & 15;
      int mhi = (lane & 16) ? 8 : 0;
#pragma unroll
      for (int v = 0; v < 8; ++v)
        rec[(mhi + v) * 96 + wave * 16 + nc] = acc[v];
    }
    __syncthreads();

    // phase C: gates for our 16x32 features
    for (int i = tid; i < B_ * 32; i += 256) {
      int bb = i >> 5, fl = i & 31, fg = g * 32 + fl;
      const float* xpt = XP + ((size_t)bb * T_ + t) * (3 * F_);
      float rz = rec[bb * 96 + fl]      + rbias[fg];
      float rr = rec[bb * 96 + 32 + fl] + rbias[F_ + fg];
      float rh = rec[bb * 96 + 64 + fl] + rbias[2 * F_ + fg];
      float z = 1.f / (1.f + __expf(-(xpt[fg] + rz)));
      float r = 1.f / (1.f + __expf(-(xpt[F_ + fg] + rr)));
      float hh = fmaxf(0.f, xpt[2 * F_ + fg] + r * rh);
      float ho = hf[bb * F_ + fg];
      float hn = z * ho + (1.f - z) * hh;
      hnxt[bb * F_ + fg] = hn;
      Gbf[((size_t)bb * T_ + t) * F_ + fg] = (bf16)hn;
    }

    // grid barrier (monotonic counter; 16 blocks always co-resident)
    __threadfence();
    __syncthreads();
    if (tid == 0) {
      __hip_atomic_fetch_add(cnt, 1u, __ATOMIC_RELEASE, __HIP_MEMORY_SCOPE_AGENT);
      unsigned target = nb * (unsigned)(t + 1);
      while (__hip_atomic_load(cnt, __ATOMIC_ACQUIRE, __HIP_MEMORY_SCOPE_AGENT) < target)
        __builtin_amdgcn_s_sleep(2);
    }
    __syncthreads();
    __threadfence();
  }
}

// ---------------------------------------------------------------------------
extern "C" void kernel_launch(void* const* d_in, const int* in_sizes, int n_in,
                              void* d_out, int out_size, void* d_ws, size_t ws_size,
                              hipStream_t stream) {
  (void)in_sizes; (void)n_in; (void)out_size; (void)ws_size;
  const float* x     = (const float*)d_in[0];
  const float* Wq    = (const float*)d_in[1];
  const float* bq    = (const float*)d_in[2];
  const float* Wk    = (const float*)d_in[3];
  const float* bk    = (const float*)d_in[4];
  const float* Wv    = (const float*)d_in[5];
  const float* bv    = (const float*)d_in[6];
  const float* Wo    = (const float*)d_in[7];
  const float* bo    = (const float*)d_in[8];
  const float* ln1g  = (const float*)d_in[9];
  const float* ln1b  = (const float*)d_in[10];
  const float* gruk  = (const float*)d_in[11];
  const float* grurk = (const float*)d_in[12];
  const float* grub  = (const float*)d_in[13];
  const float* Wd    = (const float*)d_in[14];
  const float* bd    = (const float*)d_in[15];
  const float* ln2g  = (const float*)d_in[16];
  const float* ln2b  = (const float*)d_in[17];
  float* out = (float*)d_out;

  // workspace carve-up (all sizes multiples of 256B)
  char* w = (char*)d_ws;
  size_t off = 0;
  auto carve = [&](size_t bytes) { void* p = w + off; off += (bytes + 255) & ~(size_t)255; return p; };
  const size_t W512 = (size_t)512 * 512 * sizeof(bf16);
  bf16* WQt = (bf16*)carve(W512);
  bf16* WKt = (bf16*)carve(W512);
  bf16* WVt = (bf16*)carve(W512);
  bf16* WOt = (bf16*)carve(W512);
  bf16* GKt = (bf16*)carve((size_t)1536 * 512 * sizeof(bf16));
  bf16* GRt = (bf16*)carve((size_t)1536 * 512 * sizeof(bf16));
  bf16* WDt = (bf16*)carve(W512);
  const size_t MATF = (size_t)ROWS_ * F_ * sizeof(float);   // 16 MB
  const size_t MATB = (size_t)ROWS_ * F_ * sizeof(bf16);    // 8 MB
  bf16*  XBF = (bf16*)carve(MATB);
  float* QF  = (float*)carve(MATF);
  float* KF  = (float*)carve(MATF);
  float* VF  = (float*)carve(MATF);
  float* XP  = QF;  // [8192,1536] f32 = 48 MB, reuses QF+KF+VF after q/k/v converted
  bf16*  QBF = (bf16*)carve(MATB);
  bf16*  KBF = (bf16*)carve(MATB);
  bf16*  VTB = (bf16*)carve(MATB);
  bf16*  CTX = (bf16*)carve(MATB);
  float* AF  = (float*)carve(MATF);   // attention out; reused for Dense out
  float* X1F = (float*)carve(MATF);
  bf16*  X1B = (bf16*)carve(MATB);
  bf16*  GBF = (bf16*)carve(MATB);
  float* H0  = (float*)carve((size_t)B_ * F_ * sizeof(float));
  float* H1  = (float*)carve((size_t)B_ * F_ * sizeof(float));
  unsigned* CNT = (unsigned*)carve(256);

  const size_t NX = (size_t)ROWS_ * F_;

  // init + weight prep (bf16, transposed to Bt[N,K])
  k_init<<<32, 256, 0, stream>>>(H0, H1, CNT);
  k_f32_to_bf16<<<1024, 256, 0, stream>>>(x, XBF, NX);
  k_transpose_to_bf16<<<512, 256, 0, stream>>>(Wq, WQt, 512, 512);
  k_transpose_to_bf16<<<512, 256, 0, stream>>>(Wk, WKt, 512, 512);
  k_transpose_to_bf16<<<512, 256, 0, stream>>>(Wv, WVt, 512, 512);
  k_transpose_to_bf16<<<512, 256, 0, stream>>>(Wo, WOt, 512, 512);
  k_transpose_to_bf16<<<1024, 256, 0, stream>>>(gruk, GKt, 512, 1536);
  k_transpose_to_bf16<<<1024, 256, 0, stream>>>(grurk, GRt, 512, 1536);
  k_transpose_to_bf16<<<512, 256, 0, stream>>>(Wd, WDt, 512, 512);

  // QKV projections
  dim3 g512(512 / 64, ROWS_ / 128);
  k_gemm_bf16<<<g512, 256, 0, stream>>>(XBF, WQt, bq, QF, ROWS_, 512, 512);
  k_gemm_bf16<<<g512, 256, 0, stream>>>(XBF, WKt, bk, KF, ROWS_, 512, 512);
  k_gemm_bf16<<<g512, 256, 0, stream>>>(XBF, WVt, bv, VF, ROWS_, 512, 512);
  k_f32_to_bf16<<<1024, 256, 0, stream>>>(QF, QBF, NX);
  k_f32_to_bf16<<<1024, 256, 0, stream>>>(KF, KBF, NX);
  k_transpose_v<<<1024, 256, 0, stream>>>(VF, VTB);

  // attention + output projection
  k_attn<<<dim3(T_ / 16, H_, B_), 32, 0, stream>>>(QBF, KBF, VTB, CTX);
  k_gemm_bf16<<<g512, 256, 0, stream>>>(CTX, WOt, bo, AF, ROWS_, 512, 512);

  // x1 = LN(a + x)
  k_layernorm<<<ROWS_, 256, 0, stream>>>(AF, x, ln1g, ln1b, X1F, X1B);

  // GRU: input projection then cooperative scan
  dim3 g1536(1536 / 64, ROWS_ / 128);
  k_gemm_bf16<<<g1536, 256, 0, stream>>>(X1B, GKt, grub, XP, ROWS_, 1536, 512);
  k_gru_scan<<<16, 256, 0, stream>>>(XP, GRt, grub + 1536, H0, H1, GBF, CNT);

  // Dense + final LN (AF reused for d)
  k_gemm_bf16<<<g512, 256, 0, stream>>>(GBF, WDt, bd, AF, ROWS_, 512, 512);
  k_layernorm<<<ROWS_, 256, 0, stream>>>(AF, X1F, ln2g, ln2b, out, (bf16*)nullptr);
}
